// GPT_77111842832790
// MI455X (gfx1250) — matrix operand order repaired and checked by reading, not compiled, not run on hardware
//
#include <hip/hip_runtime.h>
#include <hip/hip_bf16.h>

// ---------------------------------------------------------------------------
// CDNA5 (gfx1250) implementation: decision-transformer forward pass.
// Heavy GEMMs + attention on v_wmma_f32_16x16x32_f16 (wave32 WMMA).
// Weight tiles staged to LDS by the Tensor Data Mover (tensor_load_to_lds,
// double-buffered, s_wait_tensorcnt); activations register-pipelined b128.
// ---------------------------------------------------------------------------

typedef _Float16 v16h __attribute__((ext_vector_type(16)));
typedef _Float16 v4h  __attribute__((ext_vector_type(4)));
typedef float    v8f  __attribute__((ext_vector_type(8)));
typedef unsigned int v4u __attribute__((ext_vector_type(4)));
typedef int      v8i  __attribute__((ext_vector_type(8)));
typedef int      v4i  __attribute__((ext_vector_type(4)));

#define CFG_B   32
#define CFG_S   128
#define CFG_W   128
#define CFG_C   768
#define CFG_NH  12
#define CFG_NL  12
#define CFG_T   (CFG_W + 2 * CFG_S - 1)   // 383
#define CFG_HD  64
#define CFG_V   18

// ---------------- WMMA fragment helpers (16-bit A / B, contiguous k = 16*(lane/16) + e) --

__device__ __forceinline__ v16h frag_a_lds(const _Float16* s, int m0, int lane, int ld) {
    const _Float16* row = s + (size_t)(m0 + (lane & 15)) * ld;
    int off = (lane >> 4) ? 16 : 0;
    v16h f;
#pragma unroll
    for (int e = 0; e < 16; ++e) {
        int k = off + e;
        f[e] = row[k];
    }
    return f;
}

__device__ __forceinline__ v16h frag_b_lds(const _Float16* s, int n0, int lane, int ld) {
    const _Float16* row = s + (size_t)(n0 + (lane & 15)) * ld;
    int off = (lane >> 4) ? 16 : 0;
    v16h f;
#pragma unroll
    for (int e = 0; e < 16; ++e) f[e] = row[off + e];
    return f;
}

// B fragment straight from global K rows: B[k][n] = K[key = n][dim = k]
__device__ __forceinline__ v16h frag_b_gK(const float* base, int ldr, int key0,
                                          int koff, int lane, int T) {
    int key = key0 + (lane & 15);
    if (key > T - 1) key = T - 1;
    int off = koff + ((lane >> 4) ? 16 : 0);
    const float* row = base + (size_t)key * ldr + off;
    v16h f;
#pragma unroll
    for (int q = 0; q < 4; ++q) {
        float4 vv = *(const float4*)(row + q * 4);
        f[q * 4 + 0] = (_Float16)vv.x;
        f[q * 4 + 1] = (_Float16)vv.y;
        f[q * 4 + 2] = (_Float16)vv.z;
        f[q * 4 + 3] = (_Float16)vv.w;
    }
    return f;
}

// B fragment from global V rows: B[k][n] = V[key = k][dim = n]
__device__ __forceinline__ v16h frag_b_gV(const float* base, int ldr, int key0,
                                          int dim0, int lane, int T) {
    int dim  = dim0 + (lane & 15);
    int koff = key0 + ((lane >> 4) ? 16 : 0);
    v16h f;
#pragma unroll
    for (int e = 0; e < 16; ++e) {
        int key = koff + e;
        if (key > T - 1) key = T - 1;
        f[e] = (_Float16)base[(size_t)key * ldr + dim];
    }
    return f;
}

__device__ __forceinline__ v8f wmma_f16(v16h a, v16h b, v8f c) {
    return __builtin_amdgcn_wmma_f32_16x16x32_f16(false, a, false, b, (short)0, c,
                                                  false, false);
}

// ---------------- TDM: DMA one 128x32 f16 weight tile into LDS --------------
// D# per CDNA5 ISA. 2D tile, data_size=2B,
// tile_dim0=32 elems (64B rows), tile_dim1=128 rows, row stride = K elems.
// LDS padding: 1 DWORD after every 16 DWORDs -> LDS row pitch 34 f16.
// This toolchain exposes the 6-arg builtin: (v4u, v8i, v4i, v4i, v8i, i32).
#define WLD 34

__device__ __forceinline__ void tdm_load_w_tile(const _Float16* gsrc,
                                                unsigned ldsOff, int K, int N) {
    unsigned long long ga = (unsigned long long)(uintptr_t)gsrc;
    v4u g0;
    g0[0] = 1u;                                       // count=1, user descriptor
    g0[1] = ldsOff;                                   // lds_addr (bytes)
    g0[2] = (unsigned)(ga & 0xFFFFFFFFu);             // global_addr[31:0]
    g0[3] = (unsigned)((ga >> 32) & 0x1FFFFFFu) | (2u << 30);  // [56:32] | type=2
    v8i g1;
    g1[0] = (1 << 16)      // data_size = 1 (2 bytes)
          | (1 << 20)      // pad_enable
          | (3 << 22)      // pad_interval: 16 DWORDs (one 32-elem f16 row)
          | (0 << 25);     // pad_amount: 1 DWORD
    g1[1] = (K & 0xFFFF) << 16;                       // tensor_dim0[15:0]
    g1[2] = ((K >> 16) & 0xFFFF) | ((N & 0xFFFF) << 16);   // dim0 hi | dim1 lo
    g1[3] = ((N >> 16) & 0xFFFF) | (32 << 16);        // dim1 hi | tile_dim0=32
    g1[4] = 128;                                      // tile_dim1=128, tile_dim2=0
    g1[5] = K;                                        // tensor_dim0_stride[31:0]
    g1[6] = 0;                                        // stride0 hi | stride1 lo
    g1[7] = 0;
    v4i g2 = {0, 0, 0, 0};
    v4i g3 = {0, 0, 0, 0};
    v8i g4 = {0, 0, 0, 0, 0, 0, 0, 0};
    __builtin_amdgcn_tensor_load_to_lds(g0, g1, g2, g3, g4, 0);
}

// ---------------- Generic WMMA GEMM:  out = act(A * Wh^T + bias) [+ res] ----
// A: MxK f32 row-major. Wh: NxK f16 row-major. out/res: MxN f32.
// Block 128x128, K-step 32, 8 waves; wave tile 32x64 -> 8 wmma/step.
// A: software-pipelined register staging. W: TDM double-buffered in LDS.
#define GBM 128
#define GBN 128
#define GBK 32
#define GLD 36

template <int ACT, bool HAS_RES>
__global__ __launch_bounds__(256) void gemm_wmma(
    const float* __restrict__ A, const _Float16* __restrict__ Wh,
    const float* __restrict__ bias, const float* __restrict__ res,
    float* __restrict__ out, int M, int N, int K) {
    __shared__ __align__(16) _Float16 sA[GBM * GLD];
    __shared__ __align__(16) _Float16 sB[2][GBN * WLD];

    int tid  = threadIdx.x;
    int lane = tid & 31;
    int wid  = tid >> 5;
    int wm   = wid & 3;   // 4 waves along M (32 rows each)
    int wn   = wid >> 2;  // 2 waves along N (64 cols each)
    int bm   = blockIdx.x * GBM;
    int bn   = blockIdx.y * GBN;

    // per-thread A staging coordinates (4 float4)
    int mA[4], kA[4];
    const float* pA[4];
#pragma unroll
    for (int t = 0; t < 4; ++t) {
        int idx = t * 256 + tid;
        mA[t] = idx >> 3;
        kA[t] = (idx & 7) * 4;
        int gm = bm + mA[t];
        if (gm > M - 1) gm = M - 1;
        pA[t] = A + (size_t)gm * K + kA[t];
    }

    const _Float16* wbase = Wh + (size_t)bn * K;
    unsigned sbOff[2];
    sbOff[0] = (unsigned)(size_t)(&sB[0][0]);
    sbOff[1] = (unsigned)(size_t)(&sB[1][0]);
    bool issuer = (wid == 0);

    v8f acc[2][4];
#pragma unroll
    for (int i = 0; i < 2; ++i)
#pragma unroll
        for (int j = 0; j < 4; ++j)
#pragma unroll
            for (int e = 0; e < 8; ++e) acc[i][j][e] = 0.f;

    // prologue: W tile 0 via TDM, A tile 0 into registers
    if (issuer) tdm_load_w_tile(wbase, sbOff[0], K, N);
    float4 ra[4];
#pragma unroll
    for (int t = 0; t < 4; ++t) ra[t] = *(const float4*)(pA[t]);

    int ksteps = K / GBK;
    int buf = 0;
    for (int kb = 0; kb < ksteps; ++kb) {
        // A registers (tile kb) -> LDS
#pragma unroll
        for (int t = 0; t < 4; ++t) {
            v4h ha;
            ha[0] = (_Float16)ra[t].x; ha[1] = (_Float16)ra[t].y;
            ha[2] = (_Float16)ra[t].z; ha[3] = (_Float16)ra[t].w;
            *(v4h*)(sA + mA[t] * GLD + kA[t]) = ha;
        }
        // issue next W tile into the other buffer; wait for current tile
        if (issuer) {
            if (kb + 1 < ksteps) {
                tdm_load_w_tile(wbase + (size_t)(kb + 1) * GBK, sbOff[buf ^ 1], K, N);
                __builtin_amdgcn_s_wait_tensorcnt(1);
            } else {
                __builtin_amdgcn_s_wait_tensorcnt(0);
            }
        }
        __syncthreads();

        // issue next A tile's global loads early (hide under WMMA)
        if (kb + 1 < ksteps) {
            int k0 = (kb + 1) * GBK;
#pragma unroll
            for (int t = 0; t < 4; ++t) ra[t] = *(const float4*)(pA[t] + k0);
            if (kb + 2 < ksteps)
                __builtin_prefetch(pA[0] + (kb + 2) * GBK, 0, 1);  // global_prefetch
        }

        v16h af[2], bf[4];
#pragma unroll
        for (int i = 0; i < 2; ++i) af[i] = frag_a_lds(sA, wm * 32 + i * 16, lane, GLD);
#pragma unroll
        for (int j = 0; j < 4; ++j) bf[j] = frag_b_lds(sB[buf], wn * 64 + j * 16, lane, WLD);
#pragma unroll
        for (int i = 0; i < 2; ++i)
#pragma unroll
            for (int j = 0; j < 4; ++j)
                acc[i][j] = wmma_f16(af[i], bf[j], acc[i][j]);
        __syncthreads();
        buf ^= 1;
    }

    // epilogue (compile-time specialized)
    int half = lane >> 4;
    int nn   = lane & 15;
#pragma unroll
    for (int j = 0; j < 4; ++j) {
        int n = bn + wn * 64 + j * 16 + nn;
        float bval = bias[n];
#pragma unroll
        for (int i = 0; i < 2; ++i) {
            int mbase = bm + wm * 32 + i * 16 + 8 * half;
#pragma unroll
            for (int v = 0; v < 8; ++v) {
                int m = mbase + v;
                if (m < M) {
                    float val = acc[i][j][v] + bval;
                    if (ACT == 1) val = tanhf(val);
                    if (ACT == 2) val = 0.5f * val * (1.f + erff(val * 0.70710678f));
                    if (HAS_RES) val += res[(size_t)m * N + n];
                    out[(size_t)m * N + n] = val;
                }
            }
        }
    }
}

static inline void launch_gemm(dim3 g, hipStream_t s, const float* A, const _Float16* Wh,
                               const float* bias, const float* res, float* out,
                               int M, int N, int K, int act) {
    if (res)           gemm_wmma<0, true ><<<g, 256, 0, s>>>(A, Wh, bias, res, out, M, N, K);
    else if (act == 1) gemm_wmma<1, false><<<g, 256, 0, s>>>(A, Wh, bias, res, out, M, N, K);
    else if (act == 2) gemm_wmma<2, false><<<g, 256, 0, s>>>(A, Wh, bias, res, out, M, N, K);
    else               gemm_wmma<0, false><<<g, 256, 0, s>>>(A, Wh, bias, res, out, M, N, K);
}

// ---------------- f32 -> f16 weight conversion (vectorized) -----------------
__global__ __launch_bounds__(256) void cvt_f16_k(const float* __restrict__ src,
                                                 _Float16* __restrict__ dst,
                                                 long long n) {
    long long i = ((long long)blockIdx.x * 256 + threadIdx.x) * 4;
    if (i + 3 < n) {
        float4 vv = *(const float4*)(src + i);
        v4h hv;
        hv[0] = (_Float16)vv.x; hv[1] = (_Float16)vv.y;
        hv[2] = (_Float16)vv.z; hv[3] = (_Float16)vv.w;
        *(v4h*)(dst + i) = hv;
    }
}

// ---------------- Fused causal attention (per b, head, 16-row query tile) ---
__global__ __launch_bounds__(128) void attn_wmma(
    const float* __restrict__ qb, const float* __restrict__ kb,
    const float* __restrict__ vb, float* __restrict__ yb) {
    __shared__ __align__(16) _Float16 sQ[16 * 72];
    __shared__ float    sS[16 * 384];
    __shared__ _Float16 sP[16 * 392];

    int tid  = threadIdx.x;
    int lane = tid & 31;
    int wid  = tid >> 5;

    int qt = blockIdx.x % 24;                // 24 query tiles of 16 rows
    int hh = (blockIdx.x / 24) % CFG_NH;
    int b  = blockIdx.x / (24 * CFG_NH);
    int r0 = qt * 16;

    const float* qbase = qb + ((size_t)b * CFG_T) * CFG_C + hh * CFG_HD;
    const float* kbase = kb + ((size_t)b * CFG_T) * CFG_C + hh * CFG_HD;
    const float* vbase = vb + ((size_t)b * CFG_T) * CFG_C + hh * CFG_HD;
    float*       ybase = yb + ((size_t)b * CFG_T) * CFG_C + hh * CFG_HD;

#pragma unroll
    for (int t = 0; t < 2; ++t) {
        int idx = t * 128 + tid;
        int r = idx >> 4, kq = idx & 15;
        int gr = r0 + r;
        if (gr > CFG_T - 1) gr = CFG_T - 1;
        float4 vv = *(const float4*)(qbase + (size_t)gr * CFG_C + kq * 4);
        v4h hv;
        hv[0] = (_Float16)(vv.x * 0.125f); hv[1] = (_Float16)(vv.y * 0.125f);
        hv[2] = (_Float16)(vv.z * 0.125f); hv[3] = (_Float16)(vv.w * 0.125f);
        *(v4h*)(sQ + r * 72 + kq * 4) = hv;
    }
    __syncthreads();

    v16h a0 = frag_a_lds(sQ, 0, lane, 72);         // k 0..31
    v16h a1 = frag_a_lds(sQ + 32, 0, lane, 72);    // k 32..63
    for (int nt = wid; nt < 24; nt += 4) {
        int key0 = nt * 16;
        v8f acc;
#pragma unroll
        for (int e = 0; e < 8; ++e) acc[e] = 0.f;
        v16h bk0 = frag_b_gK(kbase, CFG_C, key0, 0, lane, CFG_T);
        v16h bk1 = frag_b_gK(kbase, CFG_C, key0, 32, lane, CFG_T);
        acc = wmma_f16(a0, bk0, acc);
        acc = wmma_f16(a1, bk1, acc);
        int key = key0 + (lane & 15);
#pragma unroll
        for (int v = 0; v < 8; ++v) {
            int m  = v + 8 * (lane >> 4);
            int gq = r0 + m;
            float sv = ((key <= gq) && (key < CFG_T)) ? acc[v] : -1e30f;
            sS[m * 384 + key] = sv;
        }
    }
    __syncthreads();

    {   // softmax: 8 threads per row, wave-local shuffle reductions
        int r = tid >> 3, p = tid & 7;
        float* row = sS + r * 384;
        float mx = -1e30f;
        for (int c = p; c < 384; c += 8) mx = fmaxf(mx, row[c]);
        mx = fmaxf(mx, __shfl_xor(mx, 1, 32));
        mx = fmaxf(mx, __shfl_xor(mx, 2, 32));
        mx = fmaxf(mx, __shfl_xor(mx, 4, 32));
        float sum = 0.f;
        for (int c = p; c < 384; c += 8) {
            float e = __expf(row[c] - mx);
            sum += e;
            row[c] = e;
        }
        sum += __shfl_xor(sum, 1, 32);
        sum += __shfl_xor(sum, 2, 32);
        sum += __shfl_xor(sum, 4, 32);
        float inv = (sum > 0.f) ? 1.f / sum : 0.f;
        for (int c = p; c < 384; c += 8)
            sP[r * 392 + c] = (_Float16)(row[c] * inv);
    }
    __syncthreads();

    {   // O = P V : each wave owns one 16-wide slice of hd
        int dim0 = wid * 16;
        v8f acc;
#pragma unroll
        for (int e = 0; e < 8; ++e) acc[e] = 0.f;
        for (int ks = 0; ks < 12; ++ks) {
            v16h pf = frag_a_lds(sP + ks * 32, 0, lane, 392);
            v16h vf = frag_b_gV(vbase, CFG_C, ks * 32, dim0, lane, CFG_T);
            acc = wmma_f16(pf, vf, acc);
        }
        int dim = dim0 + (lane & 15);
#pragma unroll
        for (int v = 0; v < 8; ++v) {
            int m  = v + 8 * (lane >> 4);
            int gr = r0 + m;
            if (gr < CFG_T) ybase[(size_t)gr * CFG_C + dim] = acc[v];
        }
    }
}

// ---------------- LayerNorm (one wave32 per row) ----------------------------
__global__ __launch_bounds__(256) void ln_rows(
    const float* __restrict__ x, const float* __restrict__ g,
    const float* __restrict__ be, float* __restrict__ out, int rows, int C) {
    int lane = threadIdx.x & 31;
    int wid  = threadIdx.x >> 5;
    int row  = blockIdx.x * 8 + wid;
    if (row >= rows) return;
    const float* xr = x + (size_t)row * C;
    float s = 0.f, s2 = 0.f;
    for (int c = lane; c < C; c += 32) { float v = xr[c]; s += v; s2 += v * v; }
#pragma unroll
    for (int m = 16; m; m >>= 1) {
        s  += __shfl_xor(s, m, 32);
        s2 += __shfl_xor(s2, m, 32);
    }
    float mean = s / C;
    float var  = s2 / C - mean * mean;
    float rstd = rsqrtf(var + 1e-5f);
    float* orow = out + (size_t)row * C;
    for (int c = lane; c < C; c += 32)
        orow[c] = (xr[c] - mean) * rstd * g[c] + be[c];
}

// ---------------- Fused conv encoder (one image per block) ------------------
__global__ __launch_bounds__(256) void conv_fused(
    const float* __restrict__ states,
    const float* __restrict__ w1, const float* __restrict__ b1,
    const float* __restrict__ w2, const float* __restrict__ b2,
    const float* __restrict__ w3, const float* __restrict__ b3,
    float* __restrict__ h3) {
    __shared__ __align__(16) _Float16 sImg[84 * 84];   // 14112 B (pre-scaled)
    __shared__ _Float16 sh1[32 * 400];                 // 25600 B
    __shared__ float    sh2[64 * 81];                  // 20736 B
    int img = blockIdx.x;
    const float* im = states + (size_t)img * 84 * 84;
    const float inv255 = 1.f / 255.f;

    for (int t = threadIdx.x; t < 1764; t += 256) {
        float4 vv = *(const float4*)(im + t * 4);
        v4h hv;
        hv[0] = (_Float16)(vv.x * inv255); hv[1] = (_Float16)(vv.y * inv255);
        hv[2] = (_Float16)(vv.z * inv255); hv[3] = (_Float16)(vv.w * inv255);
        *(v4h*)(sImg + t * 4) = hv;
    }
    __syncthreads();

    for (int o = threadIdx.x; o < 32 * 400; o += 256) {
        int ch = o / 400, rest = o % 400, y = rest / 20, xx = rest % 20;
        float sum = b1[ch];
        const float* wp = w1 + ch * 64;
#pragma unroll
        for (int i = 0; i < 8; ++i)
#pragma unroll
            for (int j = 0; j < 8; ++j)
                sum += (float)sImg[(4 * y + i) * 84 + 4 * xx + j] * wp[i * 8 + j];
        sh1[o] = (_Float16)fmaxf(sum, 0.f);
    }
    __syncthreads();
    for (int o = threadIdx.x; o < 64 * 81; o += 256) {
        int ch = o / 81, rest = o % 81, y = rest / 9, xx = rest % 9;
        float sum = b2[ch];
        for (int ic = 0; ic < 32; ++ic) {
            const float*    wp = w2 + (ch * 32 + ic) * 16;
            const _Float16* hp = sh1 + ic * 400;
#pragma unroll
            for (int i = 0; i < 4; ++i)
#pragma unroll
                for (int j = 0; j < 4; ++j)
                    sum += (float)hp[(2 * y + i) * 20 + 2 * xx + j] * wp[i * 4 + j];
        }
        sh2[o] = fmaxf(sum, 0.f);
    }
    __syncthreads();
    for (int o = threadIdx.x; o < 64 * 49; o += 256) {
        int ch = o / 49, rest = o % 49, y = rest / 7, xx = rest % 7;
        float sum = b3[ch];
        for (int ic = 0; ic < 64; ++ic) {
            const float* wp = w3 + (ch * 64 + ic) * 9;
            const float* hp = sh2 + ic * 81;
#pragma unroll
            for (int i = 0; i < 3; ++i)
#pragma unroll
                for (int j = 0; j < 3; ++j)
                    sum += hp[(y + i) * 9 + xx + j] * wp[i * 3 + j];
        }
        h3[(size_t)img * 3136 + o] = fmaxf(sum, 0.f);
    }
}

// ---------------- Token assembly ------------------------------------------
__global__ void embed_k(const int* __restrict__ actions, const int* __restrict__ text,
                        const int* __restrict__ tsteps,
                        const float* __restrict__ tok_emb, const float* __restrict__ pos_emb,
                        const float* __restrict__ gpos, const float* __restrict__ act_emb,
                        const float* __restrict__ se, float* __restrict__ x) {
    int row = blockIdx.x;               // b*T + t
    int b = row / CFG_T, t = row % CFG_T;
    int ts = tsteps[b];
    for (int c = threadIdx.x; c < CFG_C; c += blockDim.x) {
        float tokv;
        if (t < CFG_W) {
            tokv = tok_emb[(size_t)text[b * CFG_W + t] * CFG_C + c];
        } else if (((t - CFG_W) & 1) == 0) {
            tokv = se[((size_t)b * CFG_S + (t - CFG_W) / 2) * CFG_C + c];
        } else {
            int ai = (t - CFG_W + 1) / 2;
            tokv = tanhf(act_emb[(size_t)actions[b * CFG_S + ai] * CFG_C + c]);
        }
        x[(size_t)row * CFG_C + c] =
            tokv + gpos[(size_t)ts * CFG_C + c] + pos_emb[(size_t)t * CFG_C + c];
    }
}

// ---------------- Final LN + 18-way head (one wave per output row) ----------
__global__ __launch_bounds__(256) void head_k(
    const float* __restrict__ x, const float* __restrict__ g,
    const float* __restrict__ be, const float* __restrict__ hw,
    float* __restrict__ out) {
    int lane = threadIdx.x & 31;
    int wid  = threadIdx.x >> 5;
    int row  = blockIdx.x * 8 + wid;    // b*S + s
    if (row >= CFG_B * CFG_S) return;
    int b = row / CFG_S, s = row % CFG_S;
    int t = CFG_W + 2 * s;
    const float* xr = x + ((size_t)b * CFG_T + t) * CFG_C;

    float ss = 0.f, s2 = 0.f;
    for (int c = lane; c < CFG_C; c += 32) { float v = xr[c]; ss += v; s2 += v * v; }
#pragma unroll
    for (int m = 16; m; m >>= 1) {
        ss += __shfl_xor(ss, m, 32);
        s2 += __shfl_xor(s2, m, 32);
    }
    float mean = ss / CFG_C;
    float rstd = rsqrtf(s2 / CFG_C - mean * mean + 1e-5f);

    float pv[CFG_V];
#pragma unroll
    for (int v = 0; v < CFG_V; ++v) pv[v] = 0.f;
    for (int c = lane; c < CFG_C; c += 32) {
        float xn = (xr[c] - mean) * rstd * g[c] + be[c];
#pragma unroll
        for (int v = 0; v < CFG_V; ++v) pv[v] += xn * hw[v * CFG_C + c];
    }
#pragma unroll
    for (int v = 0; v < CFG_V; ++v)
#pragma unroll
        for (int m = 16; m; m >>= 1) pv[v] += __shfl_xor(pv[v], m, 32);
    if (lane == 0)
        for (int v = 0; v < CFG_V; ++v) out[(size_t)row * CFG_V + v] = pv[v];
}

// ---------------------------------------------------------------------------
extern "C" void kernel_launch(void* const* d_in, const int* in_sizes, int n_in,
                              void* d_out, int out_size, void* d_ws, size_t ws_size,
                              hipStream_t stream) {
    (void)in_sizes; (void)n_in; (void)out_size; (void)ws_size;
    const float* states  = (const float*)d_in[0];
    const int*   actions = (const int*)d_in[1];
    const int*   text    = (const int*)d_in[2];
    const int*   tsteps  = (const int*)d_in[3];
    const float* tok_emb = (const float*)d_in[4];
    const float* pos_emb = (const float*)d_in[5];
    const float* gpos    = (const float*)d_in[6];
    const float* cw1 = (const float*)d_in[7];  const float* cb1 = (const float*)d_in[8];
    const float* cw2 = (const float*)d_in[9];  const float* cb2 = (const float*)d_in[10];
    const float* cw3 = (const float*)d_in[11]; const float* cb3 = (const float*)d_in[12];
    const float* enc_w = (const float*)d_in[13];
    const float* enc_b = (const float*)d_in[14];
    const float* act_emb = (const float*)d_in[15];
    const float* ln1_g = (const float*)d_in[16]; const float* ln1_b = (const float*)d_in[17];
    const float* ln2_g = (const float*)d_in[18]; const float* ln2_b = (const float*)d_in[19];
    const float* Wq = (const float*)d_in[20]; const float* bq = (const float*)d_in[21];
    const float* Wk = (const float*)d_in[22]; const float* bk = (const float*)d_in[23];
    const float* Wv = (const float*)d_in[24]; const float* bv = (const float*)d_in[25];
    const float* Wo = (const float*)d_in[26]; const float* bo = (const float*)d_in[27];
    const float* mw1 = (const float*)d_in[28]; const float* mb1 = (const float*)d_in[29];
    const float* mw2 = (const float*)d_in[30]; const float* mb2 = (const float*)d_in[31];
    const float* lnf_g = (const float*)d_in[32]; const float* lnf_b = (const float*)d_in[33];
    const float* head_w = (const float*)d_in[34];

    const int BT   = CFG_B * CFG_T;            // 12256
    const int NIMG = CFG_B * CFG_S;            // 4096
    char* ws = (char*)d_ws;
    size_t X = (size_t)BT * CFG_C * sizeof(float);
    float* x = (float*)ws;
    float* a = (float*)(ws + X);
    float* q = (float*)(ws + 2 * X);
    float* k = (float*)(ws + 3 * X);
    float* v = (float*)(ws + 4 * X);
    float* y = (float*)(ws + 5 * X);
    float* hidden = q;                          // BT x 4C, reuses q..y region
    float* h3 = q;                              // 4096 x 3136 (pre-layer phase)
    float* se = (float*)(ws + 2 * X + (size_t)NIMG * 3136 * sizeof(float));

    // f16 weight arena after the activation slots
    const long long nEnc = (long long)CFG_C * 3136;              // 2,408,448
    const long long nQ   = (long long)CFG_NL * CFG_C * CFG_C;    // 7,077,888
    const long long nM   = (long long)CFG_NL * 4 * CFG_C * CFG_C;
    _Float16* whEnc = (_Float16*)(ws + 6 * X);
    _Float16* whQ = whEnc + nEnc;
    _Float16* whK = whQ + nQ;
    _Float16* whV = whK + nQ;
    _Float16* whO = whV + nQ;
    _Float16* whM1 = whO + nQ;
    _Float16* whM2 = whM1 + nM;

    // 0) weight f32 -> f16 conversion (one pass per call; deterministic)
    cvt_f16_k<<<(int)(nEnc / 1024), 256, 0, stream>>>(enc_w, whEnc, nEnc);
    cvt_f16_k<<<(int)(nQ / 1024), 256, 0, stream>>>(Wq, whQ, nQ);
    cvt_f16_k<<<(int)(nQ / 1024), 256, 0, stream>>>(Wk, whK, nQ);
    cvt_f16_k<<<(int)(nQ / 1024), 256, 0, stream>>>(Wv, whV, nQ);
    cvt_f16_k<<<(int)(nQ / 1024), 256, 0, stream>>>(Wo, whO, nQ);
    cvt_f16_k<<<(int)(nM / 1024), 256, 0, stream>>>(mw1, whM1, nM);
    cvt_f16_k<<<(int)(nM / 1024), 256, 0, stream>>>(mw2, whM2, nM);

    // 1) conv tower -> h3
    conv_fused<<<NIMG, 256, 0, stream>>>(states, cw1, cb1, cw2, cb2, cw3, cb3, h3);
    // 2) encoder GEMM (tanh) -> se
    {
        dim3 g((NIMG + GBM - 1) / GBM, CFG_C / GBN);
        launch_gemm(g, stream, h3, whEnc, enc_b, nullptr, se, NIMG, CFG_C, 3136, 1);
    }
    // 3) token + position assembly -> x
    embed_k<<<BT, 256, 0, stream>>>(actions, text, tsteps, tok_emb, pos_emb,
                                    gpos, act_emb, se, x);

    dim3 gP((BT + GBM - 1) / GBM, CFG_C / GBN);
    dim3 gH((BT + GBM - 1) / GBM, (4 * CFG_C) / GBN);
    int lnBlocks = (BT + 7) / 8;

    for (int l = 0; l < CFG_NL; ++l) {
        ln_rows<<<lnBlocks, 256, 0, stream>>>(x, ln1_g + l * CFG_C, ln1_b + l * CFG_C,
                                              a, BT, CFG_C);
        size_t wOff = (size_t)l * CFG_C * CFG_C;
        launch_gemm(gP, stream, a, whQ + wOff, bq + l * CFG_C, nullptr, q,
                    BT, CFG_C, CFG_C, 0);
        launch_gemm(gP, stream, a, whK + wOff, bk + l * CFG_C, nullptr, k,
                    BT, CFG_C, CFG_C, 0);
        launch_gemm(gP, stream, a, whV + wOff, bv + l * CFG_C, nullptr, v,
                    BT, CFG_C, CFG_C, 0);
        attn_wmma<<<CFG_B * CFG_NH * 24, 128, 0, stream>>>(q, k, v, y);
        launch_gemm(gP, stream, y, whO + wOff, bo + l * CFG_C, x, x,
                    BT, CFG_C, CFG_C, 0);
        ln_rows<<<lnBlocks, 256, 0, stream>>>(x, ln2_g + l * CFG_C, ln2_b + l * CFG_C,
                                              a, BT, CFG_C);
        launch_gemm(gH, stream, a, whM1 + (size_t)l * 4 * CFG_C * CFG_C,
                    mb1 + l * 4 * CFG_C, nullptr, hidden, BT, 4 * CFG_C, CFG_C, 2);
        launch_gemm(gP, stream, hidden, whM2 + (size_t)l * 4 * CFG_C * CFG_C,
                    mb2 + l * CFG_C, x, x, BT, CFG_C, 4 * CFG_C, 0);
    }

    head_k<<<(CFG_B * CFG_S + 7) / 8, 256, 0, stream>>>(x, lnf_g, lnf_b, head_w,
                                                        (float*)d_out);
}
